// GVCNN_66211215835196
// MI455X (gfx1250) — matrix operand order repaired and checked by reading, not compiled, not run on hardware
//
#include <hip/hip_runtime.h>

typedef __attribute__((ext_vector_type(16))) _Float16 v16h;
typedef __attribute__((ext_vector_type(8)))  float    v8f;
typedef __attribute__((ext_vector_type(4)))  float    v4f;

enum {
  M1 = 192,        // N*V rows
  N1 = 256,        // W1 columns
  K1 = 200704,     // D_RAW
  KSLICE = 1024,   // K depth per split-K slice
  KSPLIT = 196     // 196 * 1024 = 200704 exactly
};

// e^x without ocml: 2^(x*log2e), Taylor-6 on [-0.5,0.5], exponent via bits.
__device__ __forceinline__ float fast_exp(float x) {
  float y = x * 1.44269504088896340736f;
  float n = __builtin_floorf(y + 0.5f);
  float f = y - n;
  float p = 0.0001540353039338f;
  p = p * f + 0.0013333558146428f;
  p = p * f + 0.0096181291076285f;
  p = p * f + 0.0555041086648216f;
  p = p * f + 0.2402265069591007f;
  p = p * f + 0.6931471805599453f;
  p = p * f + 1.0f;
  union { int i; float fl; } u;
  u.i = ((int)n + 127) << 23;
  return p * u.fl;
}

// ---------------------------------------------------------------------------
// Kernel 1: LDS-free split-K GEMM. P[kb] = A(192x200704) x W1(200704x256)
// over one 1024-deep K slice. 256 threads = 8 waves; each wave owns
// 2 M-tiles x 4 N-tiles (64 accumulator VGPRs). Grid = KSPLIT * 3.
// ---------------------------------------------------------------------------
__global__ void gemm1_splitk(const float* A, const float* B, float* P) {
  const int tid  = threadIdx.x;
  const int wave = tid >> 5;
  const int lane = tid & 31;
  const int hf   = lane >> 4;
  const int l    = lane & 15;
  const int kb   = blockIdx.x / 3;                 // K-slice id
  const int pair = (blockIdx.x % 3) * 8 + wave;    // 0..23
  const int mgq  = pair >> 2;                      // 0..5 -> M-tiles 2*mgq..+1
  const int ngq  = pair & 3;                       // 0..3 -> N cols ngq*64..+63

  v8f acc[2][4];
  for (int i = 0; i < 2; ++i)
    for (int j = 0; j < 4; ++j) {
      v8f z = {0.f, 0.f, 0.f, 0.f, 0.f, 0.f, 0.f, 0.f};
      acc[i][j] = z;
    }

  for (int kk = 0; kk < KSLICE; kk += 32) {
    const long k = (long)kb * KSLICE + kk;

    // two A fragments (one per M-tile): elem i -> K = k + hf*8 + i (i<8),
    // K = k + 16 + hf*8 + (i-8) (i>=8); M = mtile*16 + l
    v16h af[2];
    for (int mt = 0; mt < 2; ++mt) {
      const float* ar = A + (long)((mgq * 2 + mt) * 16 + l) * K1 + k + hf * 8;
      v4f a0 = *(const v4f*)(ar);
      v4f a1 = *(const v4f*)(ar + 4);
      v4f a2 = *(const v4f*)(ar + 16);
      v4f a3 = *(const v4f*)(ar + 20);
      v16h t;
      t[0]  = (_Float16)a0.x; t[1]  = (_Float16)a0.y;
      t[2]  = (_Float16)a0.z; t[3]  = (_Float16)a0.w;
      t[4]  = (_Float16)a1.x; t[5]  = (_Float16)a1.y;
      t[6]  = (_Float16)a1.z; t[7]  = (_Float16)a1.w;
      t[8]  = (_Float16)a2.x; t[9]  = (_Float16)a2.y;
      t[10] = (_Float16)a2.z; t[11] = (_Float16)a2.w;
      t[12] = (_Float16)a3.x; t[13] = (_Float16)a3.y;
      t[14] = (_Float16)a3.z; t[15] = (_Float16)a3.w;
      af[mt] = t;
    }

    // four B fragments, each reused by both M-tiles:
    // elem i -> K = k + hf*8 + i (i<8), k + 16 + hf*8 + (i-8) (i>=8), N = ngq*64 + nt*16 + l
    for (int nt = 0; nt < 4; ++nt) {
      const float* bp = B + (k + hf * 8) * (long)N1 + ngq * 64 + nt * 16 + l;
      v16h bf;
      for (int i = 0; i < 16; ++i) bf[i] = (_Float16)bp[(long)(i < 8 ? i : i + 8) * N1];
      acc[0][nt] = __builtin_amdgcn_wmma_f32_16x16x32_f16(
          false, af[0], false, bf, (short)0, acc[0][nt], false, false);
      acc[1][nt] = __builtin_amdgcn_wmma_f32_16x16x32_f16(
          false, af[1], false, bf, (short)0, acc[1][nt], false, false);
    }
  }

  float* out = P + (long)kb * (M1 * N1);
  for (int mt = 0; mt < 2; ++mt)
    for (int nt = 0; nt < 4; ++nt)
      for (int r = 0; r < 8; ++r) {
        int row = (mgq * 2 + mt) * 16 + r + hf * 8;   // C/D layout
        int col = ngq * 64 + nt * 16 + l;
        out[row * N1 + col] = acc[mt][nt][r];
      }
}

// ---------------------------------------------------------------------------
// Kernel 2: deterministic split-K reduction + bias + ReLU -> h[192,256]
// ---------------------------------------------------------------------------
__global__ void reduce_bias_relu(const float* P, const float* b1, float* h) {
  const int m = blockIdx.x, t = threadIdx.x;
  float s = b1[t];
  const long off = (long)m * N1 + t;
  for (int wg = 0; wg < KSPLIT; ++wg) s += P[(long)wg * (M1 * N1) + off];
  h[m * N1 + t] = __builtin_fmaxf(s, 0.f);
}

// ---------------------------------------------------------------------------
// Kernel 3: per-sample fusion, no cross-thread communication. Every thread
// redundantly computes the 12 scores (uniform broadcast loads) and the
// per-view coefficients, then writes its own desc columns.
// desc = sum_v coeff[v]*fv[n,v,:], coeff[v)=mask*w[bin]/cnt[bin]/sum(w).
// ---------------------------------------------------------------------------
__global__ void fuse(const float* h, const float* W2, const float* b2,
                     const int* rvn, const float* fv, float* desc) {
  const int n = blockIdx.x, t = threadIdx.x;

  float score_s[12];
  for (int v = 0; v < 12; ++v) {
    const float* hr = h + (long)(n * 12 + v) * N1;
    float s = 0.f;
    for (int c = 0; c < N1; ++c) s += hr[c] * W2[c];
    float sr = __builtin_fmaxf(s + b2[0], 0.f);          // relu, >= 0
    float xs = __builtin_fminf(sr, 10.f);
    float e2 = fast_exp(2.f * xs);
    float th = (e2 - 1.f) / (e2 + 1.f);                  // tanh(|sr|)
    score_s[v] = 1.f / (1.f + fast_exp(-th));            // sigmoid
  }

  const int nv = rvn[n];
  float cnt[8] = {0, 0, 0, 0, 0, 0, 0, 0};
  int bin[12];
  for (int v = 0; v < 12; ++v) {
    int b = (int)__builtin_floorf(score_s[v] * 8.f);
    bin[v] = b < 0 ? 0 : (b > 7 ? 7 : b);
    if (v < nv) cnt[bin[v]] += 1.f;
  }
  float wg_[8], wsum = 0.f;
  for (int g = 0; g < 8; ++g) {
    float sc = __builtin_fmaxf(cnt[g], 1.f);
    float a = 0.f;
    for (int v = 0; v < 12; ++v)
      if (v < nv && bin[v] == g) a += __builtin_ceilf(score_s[v] * cnt[g]);
    float w = (cnt[g] > 0.f) ? (a / sc) : 0.f;
    wg_[g] = w;
    wsum  += w;
  }
  float coeff[12];
  for (int v = 0; v < 12; ++v)
    coeff[v] = (v < nv)
        ? wg_[bin[v]] / (__builtin_fmaxf(cnt[bin[v]], 1.f) * wsum)
        : 0.f;

  for (int d = t; d < 1024; d += 256) {
    float a = 0.f;
    for (int v = 0; v < 12; ++v)
      a += coeff[v] * fv[(long)(n * 12 + v) * 1024 + d];
    desc[n * 1024 + d] = a;
  }
}

// ---------------------------------------------------------------------------
// Kernel 4: M=16 classifier GEMM, one 16x16 WMMA tile per single-wave block.
// A: 16xK f32 (row stride K), B: KxN f32 (row stride N), C: 16xN. K % 32 == 0.
// ---------------------------------------------------------------------------
__global__ void small_gemm16(const float* A, const float* B, const float* bias,
                             float* C, int K, int N, int relu) {
  const int lane = threadIdx.x;
  const int hf   = lane >> 4;
  const int l    = lane & 15;
  const int col  = blockIdx.x * 16 + l;
  const bool cok = col < N;

  v8f acc = {0.f, 0.f, 0.f, 0.f, 0.f, 0.f, 0.f, 0.f};
  for (int k = 0; k < K; k += 32) {
    v16h af, bf;
    const float* ar = A + l * K + k;
    for (int i = 0; i < 8; ++i) {
      af[i]     = (_Float16)ar[hf * 8 + i];
      af[i + 8] = (_Float16)ar[16 + hf * 8 + i];
    }
    for (int i = 0; i < 16; ++i)
      bf[i] = cok ? (_Float16)B[(long)(k + hf * 8 + (i < 8 ? i : i + 8)) * N + col]
                  : (_Float16)0.f;
    acc = __builtin_amdgcn_wmma_f32_16x16x32_f16(
        false, af, false, bf, (short)0, acc, false, false);
  }
  for (int r = 0; r < 8; ++r) {
    int row = r + hf * 8;
    if (cok) {
      float v = acc[r] + bias[col];
      if (relu) v = __builtin_fmaxf(v, 0.f);
      C[row * N + col] = v;
    }
  }
}

// ---------------------------------------------------------------------------
extern "C" void kernel_launch(void* const* d_in, const int* in_sizes, int n_in,
                              void* d_out, int out_size, void* d_ws, size_t ws_size,
                              hipStream_t stream) {
  (void)in_sizes; (void)n_in; (void)out_size; (void)ws_size;
  const float* raw = (const float*)d_in[0];
  const float* fv  = (const float*)d_in[1];
  const int*   rvn = (const int*)  d_in[2];
  const float* W1  = (const float*)d_in[3];
  const float* b1  = (const float*)d_in[4];
  const float* W2  = (const float*)d_in[5];
  const float* b2  = (const float*)d_in[6];
  const float* Wf1 = (const float*)d_in[7];
  const float* bf1 = (const float*)d_in[8];
  const float* Wf2 = (const float*)d_in[9];
  const float* bf2 = (const float*)d_in[10];
  const float* Wl  = (const float*)d_in[11];
  const float* bl  = (const float*)d_in[12];
  float* out = (float*)d_out;

  float* ws   = (float*)d_ws;
  float* P    = ws;                                   // 196*192*256
  float* h    = P    + (size_t)KSPLIT * M1 * N1;      // 192*256
  float* desc = h    + (size_t)M1 * N1;               // 16*1024
  float* t1   = desc + 16 * 1024;                     // 16*512
  float* t2   = t1   + 16 * 512;                      // 16*256

  gemm1_splitk    <<<KSPLIT * 3, 256, 0, stream>>>(raw, W1, P);
  reduce_bias_relu<<<M1, N1, 0, stream>>>(P, b1, h);
  fuse            <<<16, 256, 0, stream>>>(h, W2, b2, rvn, fv, desc);
  small_gemm16    <<<32, 32, 0, stream>>>(desc, Wf1, bf1, t1, 1024, 512, 1);
  small_gemm16    <<<16, 32, 0, stream>>>(t1,   Wf2, bf2, t2,  512, 256, 1);
  small_gemm16    <<< 3, 32, 0, stream>>>(t2,   Wl,  bl,  out, 256,  40, 0);
}
